// RecursiveFlowS2_87625922773140
// MI455X (gfx1250) — matrix operand order repaired and checked by reading, not compiled, not run on hardware
//
#include <hip/hip_runtime.h>
#include <hip/hip_bf16.h>
#include <stdint.h>

#define B_PTS 16384
#define H_DIM 1024
#define L_LAY 8
#define NBLK  (H_DIM / 128)      // 8 n-blocks
#define NSLOT (NBLK * 2)         // 16 deterministic partial slots per layer
#define BETA_C 1.0e8f
#define EPS_C 1e-6f

typedef __attribute__((ext_vector_type(16))) __bf16 v16bf;
typedef __attribute__((ext_vector_type(8)))  float  v8f;

struct U8frag { uint4 lo; uint4 hi; };

__device__ __forceinline__ unsigned short f32_to_bf16(float f) {
    unsigned u = __builtin_bit_cast(unsigned, f);
    unsigned r = u + 0x7FFFu + ((u >> 16) & 1u);   // RNE
    return (unsigned short)(r >> 16);
}
__device__ __forceinline__ float bf16_to_f32(unsigned short s) {
    unsigned u = ((unsigned)s) << 16;
    return __builtin_bit_cast(float, u);
}

#if __has_builtin(__builtin_amdgcn_tanh_f32)
__device__ __forceinline__ float fast_tanh(float x) { return __builtin_amdgcn_tanh_f32(x); }
#elif __has_builtin(__builtin_amdgcn_tanhf)
__device__ __forceinline__ float fast_tanh(float x) { return __builtin_amdgcn_tanhf(x); }
#else
__device__ __forceinline__ float fast_tanh(float x) { return tanhf(x); }
#endif

__device__ __forceinline__ float softplus_b(float x) {
    float bx = BETA_C * x;
    if (bx > 30.0f) return x;                 // softplus saturates
    return log1pf(__expf(bx)) * (1.0f / BETA_C);
}

// lds byte-offset of a __shared__ object (generic ptr low 32 bits = LDS offset)
__device__ __forceinline__ unsigned lds_off_of(const void* p) {
    return (unsigned)(uintptr_t)p;
}
// CDNA5 async copy: 16B per lane, global -> LDS, tracked by ASYNCcnt
__device__ __forceinline__ void async_copy_b128(unsigned lds_off, unsigned long long gaddr) {
    asm volatile("global_load_async_to_lds_b128 %0, %1, off"
                 :: "v"(lds_off), "v"(gaddr) : "memory");
}

// ---------------------------------------------------------------------------
// Kernel 0: one-time W2 fp32 [l][k][n] -> bf16 transposed [l][n][k].
// Tiled through LDS so both global sides stay coalesced.
// ---------------------------------------------------------------------------
__global__ __launch_bounds__(256)
void w2cvt_kernel(const float* __restrict__ W2, unsigned short* __restrict__ W2T)
{
    __shared__ unsigned short tile[64][65];
    const int l   = blockIdx.z;
    const int k0  = blockIdx.x * 64;
    const int n0  = blockIdx.y * 64;
    const int tid = threadIdx.x;
    const int tc  = tid & 63;     // fast-moving coordinate
    const int tr  = tid >> 6;     // 0..3
    const float* src = W2 + (size_t)l * H_DIM * H_DIM;
    unsigned short* dst = W2T + (size_t)l * H_DIM * H_DIM;
#pragma unroll
    for (int r = 0; r < 16; ++r) {
        int k = tr * 16 + r;
        tile[k][tc] = f32_to_bf16(src[(size_t)(k0 + k) * H_DIM + n0 + tc]);
    }
    __syncthreads();
#pragma unroll
    for (int r = 0; r < 16; ++r) {
        int n = tr * 16 + r;
        dst[(size_t)(n0 + n) * H_DIM + k0 + tc] = tile[tc][n];
    }
}

// ---------------------------------------------------------------------------
// Kernel 1: evolve z through all 8 z-transforms; store per-layer z + log-det.
// ---------------------------------------------------------------------------
__global__ void zflow_kernel(const float* __restrict__ x,
                             const float* __restrict__ zp,
                             float* __restrict__ zpost,   // [L][B]
                             float* __restrict__ ldjz)    // [B]
{
    int i = blockIdx.x * blockDim.x + threadIdx.x;
    if (i >= B_PTS) return;
    float z = x[i * 3 + 2];
    float acc = 0.0f;
#pragma unroll
    for (int l = 0; l < L_LAY; ++l) {
        float s = zp[l * 2 + 0];
        float t = zp[l * 2 + 1];
        float u = fminf(fmaxf((z + 1.0f) * 0.5f, EPS_C), 1.0f - EPS_C);
        float logu   = __logf(u);
        float log1mu = log1pf(-u);
        float y  = __expf(s) * (logu - log1mu) + t;
        float up = 1.0f / (1.0f + __expf(-y));
        z = 2.0f * up - 1.0f;
        acc += s + __logf(up) + log1pf(-up) - logu - log1mu;
        zpost[l * B_PTS + i] = z;
    }
    ldjz[i] = acc;
}

// ---------------------------------------------------------------------------
// Kernel 2: fused MLP per layer, block tile 128x128, 8 waves, wave tile 64x32
// (4x2 frags of v_wmma_f32_16x16x32_bf16), K step 32.
//   A (h1) generated: tanh(z*W1+b1) via v_tanh_f32, staged to LDS.
//   B (W2^T bf16) streamed with double-buffered global_load_async_to_lds_b128.
//   Epilogue: tanh(+b2) -> LDS, per-half-row dot with W3 -> deterministic
//   partial slot (no atomics).
// ---------------------------------------------------------------------------
union SharedU {
    struct {
        unsigned short A[128][32];        // 8 KB  : h1 tile [m][k]
        unsigned short Bb[2][128][32];    // 16 KB : W2^T tiles [n][k], double buffered
    } s;
    unsigned short h2[128][128];          // 32 KB : epilogue tile (aliased)
};

__global__ __launch_bounds__(256)
void mlp_gemm_kernel(const float* __restrict__ zpost,
                     const float* __restrict__ W1, const float* __restrict__ b1,
                     const unsigned short* __restrict__ W2T,
                     const float* __restrict__ b2,
                     const float* __restrict__ W3,
                     float* __restrict__ ppart)   // [L][NSLOT][B][2]
{
    __shared__ __align__(16) SharedU sh;
    __shared__ float zls[128];
    __shared__ float b2s[128];
    __shared__ float w3s[128][2];

    const int l    = blockIdx.z;
    const int m0   = blockIdx.x * 128;
    const int nb   = blockIdx.y;
    const int n0   = nb * 128;
    const int tid  = threadIdx.x;
    const int lane = tid & 31;
    const int wave = tid >> 5;
    const int wm0  = (wave >> 2) * 64;
    const int wn0  = (wave & 3) * 32;
    const int half = lane >> 4;
    const int l15  = lane & 15;

    const float* W1l = W1 + (size_t)l * H_DIM;
    const float* b1l = b1 + (size_t)l * H_DIM;
    // byte base of this block's W2^T panel: rows n0..n0+127, 2048 B per row
    const unsigned long long gB =
        (unsigned long long)(uintptr_t)(W2T + ((size_t)l * H_DIM + n0) * H_DIM);

    if (tid < 128) {
        zls[tid] = zpost[l * B_PTS + m0 + tid];
        b2s[tid] = b2[l * H_DIM + n0 + tid];
    }
    ((float*)w3s)[tid] = W3[(size_t)l * H_DIM * 2 + n0 * 2 + tid];

    // per-thread async chunk coordinates: chunk c covers LDS bytes [c*16, c*16+16)
    // of a [128][32]-bf16 tile (64 B per n-row -> n = c>>2, kbyte = (c&3)*16)
    const int c0 = tid;          // chunks 0..255
    const int c1 = tid + 256;    // chunks 256..511
    const unsigned bB0 = lds_off_of(&sh.s.Bb[0][0][0]);
    const unsigned bB1 = lds_off_of(&sh.s.Bb[1][0][0]);

    // prologue: async-fill buffer 0 with k-chunk 0
    {
        unsigned long long g = gB;   // kc = 0
        async_copy_b128(bB0 + c0 * 16, g + (unsigned)(c0 >> 2) * 2048u + (unsigned)(c0 & 3) * 16u);
        async_copy_b128(bB0 + c1 * 16, g + (unsigned)(c1 >> 2) * 2048u + (unsigned)(c1 & 3) * 16u);
    }

    v8f acc[4][2];
#pragma unroll
    for (int mi = 0; mi < 4; ++mi)
#pragma unroll
        for (int ni = 0; ni < 2; ++ni)
#pragma unroll
            for (int r = 0; r < 8; ++r) acc[mi][ni][r] = 0.0f;

    const int sm = tid >> 1;          // A staging row m, 0..127
    const int sk = (tid & 1) * 16;    // A staging k offset

    int p = 0;
    for (int kc = 0; kc < H_DIM; kc += 32) {
        __syncthreads();
        // ---- stage A tile: h1 = tanh(z * W1 + b1) (generated, bf16 [m][k]) ----
        {
            float zm = zls[sm];
            const float4* w4 = (const float4*)(W1l + kc + sk);
            const float4* c4 = (const float4*)(b1l + kc + sk);
            unsigned short tmp[16];
#pragma unroll
            for (int q = 0; q < 4; ++q) {
                float4 w = w4[q];
                float4 c = c4[q];
                tmp[q * 4 + 0] = f32_to_bf16(fast_tanh(fmaf(zm, w.x, c.x)));
                tmp[q * 4 + 1] = f32_to_bf16(fast_tanh(fmaf(zm, w.y, c.y)));
                tmp[q * 4 + 2] = f32_to_bf16(fast_tanh(fmaf(zm, w.z, c.z)));
                tmp[q * 4 + 3] = f32_to_bf16(fast_tanh(fmaf(zm, w.w, c.w)));
            }
            *(uint4*)&sh.s.A[sm][sk]     = *(uint4*)&tmp[0];
            *(uint4*)&sh.s.A[sm][sk + 8] = *(uint4*)&tmp[8];
        }
        // ---- async-prefetch next W2^T tile into the other buffer ----
        if (kc + 32 < H_DIM) {
            unsigned bnext = p ? bB0 : bB1;
            unsigned long long g = gB + (unsigned)(kc + 32) * 2u;
            async_copy_b128(bnext + c0 * 16, g + (unsigned)(c0 >> 2) * 2048u + (unsigned)(c0 & 3) * 16u);
            async_copy_b128(bnext + c1 * 16, g + (unsigned)(c1 >> 2) * 2048u + (unsigned)(c1 & 3) * 16u);
            asm volatile("s_wait_asynccnt 0x2" ::: "memory");  // current buffer done
        } else {
            asm volatile("s_wait_asynccnt 0x0" ::: "memory");
        }
        __syncthreads();

        // ---- B frags: lanes 0-15: N=lane, K={0..7,16..23}; lanes 16-31: K={8..15,24..31} ----
        v16bf bfrag[2];
#pragma unroll
        for (int ni = 0; ni < 2; ++ni) {
            int n  = wn0 + ni * 16 + l15;
            int ko = half * 8;
            U8frag t;
            t.lo = *(const uint4*)&sh.s.Bb[p][n][ko];
            t.hi = *(const uint4*)&sh.s.Bb[p][n][ko + 16];
            bfrag[ni] = __builtin_bit_cast(v16bf, t);
        }
        // ---- A frags + WMMA: lanes 0-15: K={0..7,16..23}; 16-31: K={8..15,24..31}
#pragma unroll
        for (int mi = 0; mi < 4; ++mi) {
            int m  = wm0 + mi * 16 + l15;
            int k1 = half * 8;
            U8frag t;
            t.lo = *(const uint4*)&sh.s.A[m][k1];
            t.hi = *(const uint4*)&sh.s.A[m][k1 + 16];
            v16bf afrag = __builtin_bit_cast(v16bf, t);
#pragma unroll
            for (int ni = 0; ni < 2; ++ni) {
                acc[mi][ni] = __builtin_amdgcn_wmma_f32_16x16x32_bf16(
                    false, afrag, false, bfrag[ni], (short)0, acc[mi][ni],
                    false, false);
            }
        }
        p ^= 1;
    }

    // ---- epilogue: h2 = tanh(acc + b2) -> LDS (aliased) ----
    __syncthreads();
#pragma unroll
    for (int mi = 0; mi < 4; ++mi) {
#pragma unroll
        for (int ni = 0; ni < 2; ++ni) {
            int fn = wn0 + ni * 16 + l15;
#pragma unroll
            for (int r = 0; r < 8; ++r) {
                int fm = wm0 + mi * 16 + half * 8 + r;   // C layout: M = half*8 + r
                float v = acc[mi][ni][r] + b2s[fn];
                sh.h2[fm][fn] = f32_to_bf16(fast_tanh(v));
            }
        }
    }
    __syncthreads();

    // ---- p partial: all 256 threads; thread owns (row, n-half of 64) ----
    {
        const int row  = tid & 127;
        const int nh   = tid >> 7;            // 0 or 1
        const int base = nh * 64;
        float s0 = 0.0f, s1 = 0.0f;
#pragma unroll 8
        for (int n = 0; n < 64; ++n) {
            float h = bf16_to_f32(sh.h2[row][base + n]);
            s0 = fmaf(h, w3s[base + n][0], s0);
            s1 = fmaf(h, w3s[base + n][1], s1);
        }
        // deterministic slot: [l][nb*2+nh][m][2]  (plain store, no atomics)
        size_t slot = ((size_t)l * NSLOT + (nb * 2 + nh)) * B_PTS + (m0 + row);
        ppart[slot * 2 + 0] = s0;
        ppart[slot * 2 + 1] = s1;
    }
}

// ---------------------------------------------------------------------------
// Kernel 3: deterministic p reduction + serial Mobius/rotation chain.
// ---------------------------------------------------------------------------
__global__ void mobius_kernel(const float* __restrict__ x,
                              const float* __restrict__ rot,
                              const float* __restrict__ b3,
                              const float* __restrict__ ppart,  // [L][NSLOT][B][2]
                              const float* __restrict__ zpost,
                              const float* __restrict__ ldjz,
                              float* __restrict__ out)
{
    int i = blockIdx.x * blockDim.x + threadIdx.x;
    if (i >= B_PTS) return;
    float x0 = x[i * 3 + 0], x1 = x[i * 3 + 1], z0 = x[i * 3 + 2];
    float rho = sqrtf(softplus_b(1.0f - z0 * z0));
    float xa = x0 / rho, xb = x1 / rho;
    float ldv = 0.0f;
#pragma unroll
    for (int l = 0; l < L_LAY; ++l) {
        float p0 = b3[l * 2 + 0];
        float p1 = b3[l * 2 + 1];
#pragma unroll
        for (int sl = 0; sl < NSLOT; ++sl) {    // fixed order => deterministic
            size_t slot = ((size_t)l * NSLOT + sl) * B_PTS + i;
            p0 += ppart[slot * 2 + 0];
            p1 += ppart[slot * 2 + 1];
        }
        float inv = 1.0f / (1.0f + sqrtf(p0 * p0 + p1 * p1));
        float w0 = p0 * inv, w1 = p1 * inv;
        float d0 = xa - w0, d1 = xb - w1;
        float d2 = d0 * d0 + d1 * d1;
        float fac = (1.0f - (w0 * w0 + w1 * w1)) / d2;
        float o0 = fac * d0 - w0, o1 = fac * d1 - w1;
        ldv += __logf(fac);
        float th = rot[l];
        float c, s;
        __sincosf(th, &s, &c);
        xa = c * o0 - s * o1;
        xb = s * o0 + c * o1;
    }
    float zf = zpost[(L_LAY - 1) * B_PTS + i];
    float rhof = sqrtf(softplus_b(1.0f - zf * zf));
    out[i * 3 + 0] = xa * rhof;
    out[i * 3 + 1] = xb * rhof;
    out[i * 3 + 2] = zf;
    out[B_PTS * 3 + i] = ldv + ldjz[i];
}

// ---------------------------------------------------------------------------
extern "C" void kernel_launch(void* const* d_in, const int* in_sizes, int n_in,
                              void* d_out, int out_size, void* d_ws, size_t ws_size,
                              hipStream_t stream)
{
    (void)in_sizes; (void)n_in; (void)out_size; (void)ws_size;
    const float* x   = (const float*)d_in[0];
    const float* zp  = (const float*)d_in[1];
    const float* rot = (const float*)d_in[2];
    const float* W1  = (const float*)d_in[3];
    const float* b1  = (const float*)d_in[4];
    const float* W2  = (const float*)d_in[5];
    const float* b2  = (const float*)d_in[6];
    const float* W3  = (const float*)d_in[7];
    const float* b3  = (const float*)d_in[8];
    float* out = (float*)d_out;

    float* zpost = (float*)d_ws;                              // L*B f32
    float* ldjz  = zpost + (size_t)L_LAY * B_PTS;             // B f32
    float* ppart = ldjz + B_PTS;                              // L*NSLOT*B*2 f32
    unsigned short* W2T =
        (unsigned short*)(ppart + (size_t)L_LAY * NSLOT * B_PTS * 2); // L*H*H bf16

    dim3 gcvt(H_DIM / 64, H_DIM / 64, L_LAY);
    w2cvt_kernel<<<gcvt, 256, 0, stream>>>(W2, W2T);

    zflow_kernel<<<B_PTS / 256, 256, 0, stream>>>(x, zp, zpost, ldjz);

    dim3 grid(B_PTS / 128, NBLK, L_LAY);
    mlp_gemm_kernel<<<grid, 256, 0, stream>>>(zpost, W1, b1, W2T, b2, W3, ppart);

    mobius_kernel<<<B_PTS / 256, 256, 0, stream>>>(x, rot, b3, ppart, zpost, ldjz, out);
}
